// NonLocal3D_7481833029823
// MI455X (gfx1250) — matrix operand order repaired and checked by reading, not compiled, not run on hardware
//
#include <hip/hip_runtime.h>
#include <hip/hip_bf16.h>

typedef __attribute__((ext_vector_type(16))) __bf16 v16bf;
typedef __attribute__((ext_vector_type(8)))  float  v8f;

#define M_TOT 6272   // 8*28*28
#define C_IN  512
#define H_CH  256
#define BATCH 2
#define QT 64        // query rows per attention block
#define KT 64        // keys per inner step

#define WMMA_BF16(a, b, c) \
  __builtin_amdgcn_wmma_f32_16x16x32_bf16(false, (a), false, (b), (short)0, (c), false, false)

union frag_u { v16bf v; uint4 q[2]; };
union bf8_u  { uint4 q; __bf16 h[8]; };

// A: 16x32 bf16 tile, source row-major [16][>=32], row stride ld (elems).
// Per lane: two contiguous 16B runs -> 2x ds_load_b128.
__device__ __forceinline__ v16bf load_a_frag(const __bf16* p, int ld) {
  const int lane = threadIdx.x & 31;
  const __bf16* row = p + (lane & 15) * ld + ((lane < 16) ? 0 : 8);
  frag_u u;
  u.q[0] = *(const uint4*)row;
  u.q[1] = *(const uint4*)(row + 16);
  return u.v;
}

// B: logical [K=32][N=16], stored TRANSPOSED [N][K] row-major, stride ld (elems).
// Per lane: two contiguous 16B runs -> 2x ds_load_b128.
__device__ __forceinline__ v16bf load_bt_frag(const __bf16* p, int ld) {
  const int lane = threadIdx.x & 31;
  const __bf16* row = p + (lane & 15) * ld + ((lane < 16) ? 0 : 8);
  frag_u u;
  u.q[0] = *(const uint4*)row;
  u.q[1] = *(const uint4*)(row + 16);
  return u.v;
}

// 8 contiguous fp32 (16B-aligned) -> 8 bf16, single 16B LDS store.
__device__ __forceinline__ void cvt8_store(__bf16* dst, const float* src) {
  const float4 f0 = ((const float4*)src)[0];
  const float4 f1 = ((const float4*)src)[1];
  bf8_u u;
  u.h[0] = (__bf16)f0.x; u.h[1] = (__bf16)f0.y;
  u.h[2] = (__bf16)f0.z; u.h[3] = (__bf16)f0.w;
  u.h[4] = (__bf16)f1.x; u.h[5] = (__bf16)f1.y;
  u.h[6] = (__bf16)f1.z; u.h[7] = (__bf16)f1.w;
  *(uint4*)dst = u.q;
}

// ---------------------------------------------------------------------------
// Kernel 1: projections. out[m][o] = sum_c x[c][m] * w[o][c], bf16 out.
// Grid: (M/64, 768/64, B).
// ---------------------------------------------------------------------------
__global__ __launch_bounds__(256)
void nl3d_proj(const float* __restrict__ x,
               const float* __restrict__ w_theta,
               const float* __restrict__ w_phi,
               const float* __restrict__ w_g,
               __bf16* __restrict__ theta,
               __bf16* __restrict__ phi,
               __bf16* __restrict__ g) {
  const int mb = blockIdx.x;
  const int ob = blockIdx.y;
  const int n  = blockIdx.z;
  const int tid = threadIdx.x;
  const int wave = tid >> 5, lane = tid & 31;

  const int o0 = ob * 64;
  const float* w; __bf16* outp; int ol0;
  if (o0 < 256)      { w = w_theta; outp = theta; ol0 = o0; }
  else if (o0 < 512) { w = w_phi;   outp = phi;   ol0 = o0 - 256; }
  else               { w = w_g;     outp = g;     ol0 = o0 - 512; }

  __shared__ alignas(16) __bf16 sh[2 * 64 * 32];  // At | BtT, reused as 64x64 tile
  __bf16* At  = sh;            // A[m][c] (x transposed at fill)
  __bf16* BtT = sh + 64 * 32;  // B^T: [o][c] (contiguous from w)

  v8f acc[2] = {};
  const int wr = (wave & 3) * 16;
  const int wc = (wave >> 2) * 16;     // second tile at +32

  const size_t xbase = (size_t)n * C_IN * M_TOT + (size_t)mb * 64;
  for (int c0 = 0; c0 < C_IN; c0 += 32) {
    {  // A[m][c] = x[c][m]: scalar gather along c, vector 16B store per thread
      const int mm = tid & 63;
      const int cb = (tid >> 6) * 8;
      float t8[8];
#pragma unroll
      for (int r = 0; r < 8; ++r)
        t8[r] = x[xbase + (size_t)(c0 + cb + r) * M_TOT + mm];
      bf8_u u;
#pragma unroll
      for (int r = 0; r < 8; ++r) u.h[r] = (__bf16)t8[r];
      *(uint4*)&At[mm * 32 + cb] = u.q;
      if (c0 + 32 < C_IN)  // hint next c-block (global_prefetch_b8)
        __builtin_prefetch(&x[xbase + (size_t)(c0 + 32 + cb) * M_TOT + mm], 0, 0);
    }
    {  // B^T[o][c] = w[o][c]: contiguous rows, vector convert
      const int oo  = tid >> 2;
      const int cc8 = (tid & 3) * 8;
      cvt8_store(&BtT[oo * 32 + cc8], &w[(ol0 + oo) * C_IN + c0 + cc8]);
    }
    __syncthreads();
    v16bf a = load_a_frag(&At[wr * 32], 32);
#pragma unroll
    for (int j = 0; j < 2; ++j) {
      v16bf b = load_bt_frag(&BtT[(wc + j * 32) * 32], 32);
      acc[j] = WMMA_BF16(a, b, acc[j]);
    }
    __syncthreads();
  }

  // Stage 64x64 bf16 tile in LDS, then coalesced b128 stores.
  const int rowoff = (lane < 16) ? 0 : 8;
  const int coll = lane & 15;
  __bf16* tile = sh;  // 64x64
#pragma unroll
  for (int j = 0; j < 2; ++j)
#pragma unroll
    for (int i = 0; i < 8; ++i)
      tile[(wr + i + rowoff) * 64 + wc + j * 32 + coll] = (__bf16)acc[j][i];
  __syncthreads();
  for (int i = tid; i < 64 * 64 / 8; i += 256) {
    const int mm = i >> 3, seg = i & 7;
    *(uint4*)&outp[((size_t)n * M_TOT + mb * 64 + mm) * H_CH + ol0 + seg * 8] =
        *(const uint4*)&tile[mm * 64 + seg * 8];
  }
}

// ---------------------------------------------------------------------------
// Kernel 2: flash attention. scale = 16 (faithful *sqrt(d)); fp32 online
// softmax; software-pipelined K/V tiles.  Grid: (M/QT, B).
// ---------------------------------------------------------------------------
#define ATTN_SMEM (QT * H_CH * 2 + KT * H_CH * 2 + H_CH * KT * 2 + \
                   QT * KT * 4 + QT * KT * 2 + 3 * QT * 4)

__global__ __launch_bounds__(256)
void nl3d_attn(const __bf16* __restrict__ theta, const __bf16* __restrict__ phi,
               const __bf16* __restrict__ g, __bf16* __restrict__ y) {
  const int qb = blockIdx.x;
  const int n  = blockIdx.y;
  const int tid = threadIdx.x;
  const int wave = tid >> 5, lane = tid & 31;

  extern __shared__ char smem[];
  __bf16* Qt  = (__bf16*)smem;              // [QT][H_CH]; reused to stage y
  __bf16* Kt  = Qt + QT * H_CH;             // [KT][H_CH]
  __bf16* VtT = Kt + KT * H_CH;             // [H_CH][KT]  (V transposed)
  float*  Sf  = (float*)(VtT + H_CH * KT);  // [QT][KT] fp32 scores
  __bf16* Pt  = (__bf16*)(Sf + QT * KT);    // [QT][KT] bf16 probabilities
  float* rowm = (float*)(Pt + QT * KT);
  float* rowl = rowm + QT;
  float* rowf = rowl + QT;

  const __bf16* Qg = theta + ((size_t)n * M_TOT + (size_t)qb * QT) * H_CH;
  for (int i = tid; i < QT * H_CH / 8; i += 256)
    ((uint4*)Qt)[i] = ((const uint4*)Qg)[i];
  if (tid < QT) { rowm[tid] = -1e30f; rowl[tid] = 0.0f; }

  const int wr  = (wave & 3) * 16;     // query-row group
  const int wch = (wave >> 2) * 128;   // d-column half (PV)
  const int sc  = (wave >> 2) * 16;    // key-column group (S): sc and sc+32
  const int rowoff = (lane < 16) ? 0 : 8;
  const int coll = lane & 15;

  v8f acc[8] = {};
  const int nkb = M_TOT / KT;

  // Pipeline prologue: tile 0 -> registers.
  uint4 kreg[8], vreg[8];
  {
    const uint4* Kg = (const uint4*)(phi + (size_t)n * M_TOT * H_CH);
    const uint4* Vg = (const uint4*)(g   + (size_t)n * M_TOT * H_CH);
#pragma unroll
    for (int r = 0; r < 8; ++r) {
      kreg[r] = Kg[tid + 256 * r];
      vreg[r] = Vg[tid + 256 * r];
    }
  }
  __syncthreads();

  for (int kb = 0; kb < nkb; ++kb) {
    // Commit registers -> LDS (K kept [key][d]; V transposed -> [d][key]).
#pragma unroll
    for (int r = 0; r < 8; ++r) {
      const int i = tid + 256 * r;
      ((uint4*)Kt)[i] = kreg[r];
      bf8_u u; u.q = vreg[r];
      const int key = i >> 5;
      const int d0  = (i & 31) * 8;
#pragma unroll
      for (int j = 0; j < 8; ++j) VtT[(d0 + j) * KT + key] = u.h[j];
    }
    __syncthreads();

    // Issue next tile's global loads now; they overlap S + softmax + PV.
    if (kb + 1 < nkb) {
      const uint4* Kg =
          (const uint4*)(phi + ((size_t)n * M_TOT + (size_t)(kb + 1) * KT) * H_CH);
      const uint4* Vg =
          (const uint4*)(g + ((size_t)n * M_TOT + (size_t)(kb + 1) * KT) * H_CH);
#pragma unroll
      for (int r = 0; r < 8; ++r) {
        kreg[r] = Kg[tid + 256 * r];
        vreg[r] = Vg[tid + 256 * r];
      }
    }

    // S = (Q @ K^T) * 16 ; two 16x16 tiles per wave of the 64x64 block
    v8f s[2] = {};
#pragma unroll
    for (int d0 = 0; d0 < H_CH; d0 += 32) {
      v16bf a = load_a_frag(Qt + wr * H_CH + d0, H_CH);
#pragma unroll
      for (int jj = 0; jj < 2; ++jj) {
        v16bf b = load_bt_frag(Kt + (sc + jj * 32) * H_CH + d0, H_CH);
        s[jj] = WMMA_BF16(a, b, s[jj]);
      }
    }
#pragma unroll
    for (int jj = 0; jj < 2; ++jj)
#pragma unroll
      for (int i = 0; i < 8; ++i)
        Sf[(wr + i + rowoff) * KT + sc + jj * 32 + coll] = s[jj][i] * 16.0f;
    __syncthreads();

    // fp32 online softmax: 4 threads per row (16 keys each), shfl reductions
    {
      const int r = tid >> 2;
      const int qq = tid & 3;
      const float* srow = &Sf[r * KT + qq * 16];
      float sv[16];
#pragma unroll
      for (int v4 = 0; v4 < 4; ++v4) {
        const float4 f = ((const float4*)srow)[v4];
        sv[4 * v4 + 0] = f.x; sv[4 * v4 + 1] = f.y;
        sv[4 * v4 + 2] = f.z; sv[4 * v4 + 3] = f.w;
      }
      float mx = sv[0];
#pragma unroll
      for (int k = 1; k < 16; ++k) mx = fmaxf(mx, sv[k]);
      mx = fmaxf(mx, __shfl_xor(mx, 1, 4));
      mx = fmaxf(mx, __shfl_xor(mx, 2, 4));
      mx = fmaxf(mx, rowm[r]);
      float sum = 0.0f;
      bf8_u u0, u1;
#pragma unroll
      for (int k = 0; k < 8; ++k) {
        const float p = __expf(sv[k] - mx);
        u0.h[k] = (__bf16)p;
        sum += p;
      }
#pragma unroll
      for (int k = 0; k < 8; ++k) {
        const float p = __expf(sv[8 + k] - mx);
        u1.h[k] = (__bf16)p;
        sum += p;
      }
      *(uint4*)&Pt[r * KT + qq * 16]     = u0.q;
      *(uint4*)&Pt[r * KT + qq * 16 + 8] = u1.q;
      sum += __shfl_xor(sum, 1, 4);
      sum += __shfl_xor(sum, 2, 4);
      if (qq == 0) {
        const float f = __expf(rowm[r] - mx);
        rowl[r] = rowl[r] * f + sum;
        rowm[r] = mx;
        rowf[r] = f;
      }
    }
    __syncthreads();

    // acc = acc * f + P @ V  (two A frags chain over 64 keys)
    float fr[8];
#pragma unroll
    for (int i = 0; i < 8; ++i) fr[i] = rowf[wr + i + rowoff];
    v16bf a0 = load_a_frag(Pt + wr * KT, KT);
    v16bf a1 = load_a_frag(Pt + wr * KT + 32, KT);
#pragma unroll
    for (int j = 0; j < 8; ++j) {
#pragma unroll
      for (int i = 0; i < 8; ++i) acc[j][i] *= fr[i];
      v16bf b0 = load_bt_frag(&VtT[(wch + j * 16) * KT], KT);
      acc[j] = WMMA_BF16(a0, b0, acc[j]);
      v16bf b1 = load_bt_frag(&VtT[(wch + j * 16) * KT + 32], KT);
      acc[j] = WMMA_BF16(a1, b1, acc[j]);
    }
    __syncthreads();
  }

  // Normalize, stage y tile in (now-dead) Qt LDS, coalesced b128 stores.
  float inv[8];
#pragma unroll
  for (int i = 0; i < 8; ++i) inv[i] = 1.0f / rowl[wr + i + rowoff];
#pragma unroll
  for (int j = 0; j < 8; ++j)
#pragma unroll
    for (int i = 0; i < 8; ++i)
      Qt[(wr + i + rowoff) * H_CH + wch + j * 16 + coll] =
          (__bf16)(acc[j][i] * inv[i]);
  __syncthreads();
  __bf16* Yg = y + ((size_t)n * M_TOT + (size_t)qb * QT) * H_CH;
  for (int i = tid; i < QT * H_CH / 8; i += 256)
    ((uint4*)Yg)[i] = ((const uint4*)Qt)[i];
}

// ---------------------------------------------------------------------------
// Kernel 3: out[c][m] = x[c][m] + sum_o w_out[c][o] * y[m][o].  fp32 output.
// Grid: (M/64, C/64, B).  Output stores are naturally coalesced (consecutive m).
// ---------------------------------------------------------------------------
__global__ __launch_bounds__(256)
void nl3d_out(const float* __restrict__ x, const float* __restrict__ w_out,
              const __bf16* __restrict__ y, float* __restrict__ out) {
  const int mb  = blockIdx.x;
  const int cbk = blockIdx.y;
  const int n   = blockIdx.z;
  const int tid = threadIdx.x;
  const int wave = tid >> 5, lane = tid & 31;

  __shared__ alignas(16) __bf16 At[64 * 32];    // A[c][o]  (contiguous from w_out)
  __shared__ alignas(16) __bf16 BtT[64 * 32];   // B^T: [m][o] == y native layout

  v8f acc[2] = {};
  const int wr = (wave & 3) * 16;
  const int wc = (wave >> 2) * 16;

  for (int o0 = 0; o0 < H_CH; o0 += 32) {
    {
      const int rr  = tid >> 2;
      const int ob8 = (tid & 3) * 8;
      cvt8_store(&At[rr * 32 + ob8], &w_out[(cbk * 64 + rr) * H_CH + o0 + ob8]);
    }
    {
      const int mm  = tid >> 2;
      const int ob8 = (tid & 3) * 8;
      *(uint4*)&BtT[mm * 32 + ob8] =
          *(const uint4*)&y[((size_t)n * M_TOT + mb * 64 + mm) * H_CH + o0 + ob8];
    }
    __syncthreads();
    v16bf a = load_a_frag(&At[wr * 32], 32);
#pragma unroll
    for (int j = 0; j < 2; ++j) {
      v16bf b = load_bt_frag(&BtT[(wc + j * 32) * 32], 32);
      acc[j] = WMMA_BF16(a, b, acc[j]);
    }
    __syncthreads();
  }

  const int rowoff = (lane < 16) ? 0 : 8;
  const int coll = lane & 15;
#pragma unroll
  for (int j = 0; j < 2; ++j)
#pragma unroll
    for (int i = 0; i < 8; ++i) {
      const int c = cbk * 64 + wr + i + rowoff;
      const int m = mb * 64 + wc + j * 32 + coll;
      const size_t idx = ((size_t)n * C_IN + c) * M_TOT + m;
      out[idx] = x[idx] + acc[j][i];
    }
}

// ---------------------------------------------------------------------------
extern "C" void kernel_launch(void* const* d_in, const int* in_sizes, int n_in,
                              void* d_out, int out_size, void* d_ws, size_t ws_size,
                              hipStream_t stream) {
  (void)in_sizes; (void)n_in; (void)out_size; (void)ws_size;
  const float* x       = (const float*)d_in[0];
  const float* w_g     = (const float*)d_in[1];
  const float* w_theta = (const float*)d_in[2];
  const float* w_phi   = (const float*)d_in[3];
  const float* w_out   = (const float*)d_in[4];
  float* out = (float*)d_out;

  // workspace: theta | phi | g | y   (bf16, [B][M][H_CH] each) ~= 25.7 MB
  const size_t T = (size_t)BATCH * M_TOT * H_CH;
  __bf16* theta = (__bf16*)d_ws;
  __bf16* phi   = theta + T;
  __bf16* g     = phi + T;
  __bf16* y     = g + T;

  const dim3 blk(256);
  nl3d_proj<<<dim3(M_TOT / 64, (3 * H_CH) / 64, BATCH), blk, 0, stream>>>(
      x, w_theta, w_phi, w_g, theta, phi, g);
  nl3d_attn<<<dim3(M_TOT / QT, BATCH), blk, ATTN_SMEM, stream>>>(theta, phi, g, y);
  nl3d_out<<<dim3(M_TOT / 64, C_IN / 64, BATCH), blk, 0, stream>>>(x, w_out, y, out);
}
